// CRF_77455440216700
// MI455X (gfx1250) — matrix operand order repaired and checked by reading, not compiled, not run on hardware
//
#include <hip/hip_runtime.h>
#include <hip/hip_bf16.h>
#include <stdint.h>

#define Bn 512
#define Sn 512
#define Tn 128
#define TB 16   // batch tile per workgroup (WMMA M)

#define LOG2E 1.4426950408889634f
#define LN2   0.6931471805599453f

typedef _Float16 v16h __attribute__((ext_vector_type(16)));
typedef _Float16 h8   __attribute__((ext_vector_type(8)));
typedef float    v8f  __attribute__((ext_vector_type(8)));
typedef uint32_t v4u  __attribute__((ext_vector_type(4)));
typedef int      v8i  __attribute__((ext_vector_type(8)));
typedef int      v4i  __attribute__((ext_vector_type(4)));

__device__ __forceinline__ uint32_t lds_off_u32(const void* p) {
  // generic LDS pointer: low 32 bits are the LDS byte offset
  return (uint32_t)(uintptr_t)p;
}

// Volatile LDS read via explicit addrspace(3): lowers to ds_load_b32 and
// cannot be folded away even though the TDM writes are invisible to the
// compiler.
__device__ __forceinline__ float lds_load_f32_vol(uint32_t off) {
  typedef volatile __attribute__((address_space(3))) float vlds_t;
  return *(vlds_t*)off;
}

// TDM: load a 16(row,batch-strided) x 128(contig f32) tile into LDS.
// D# per CDNA5 ISA (2-D tensor, groups 2/3 zero).
// clang-23 toolchain: 6-arg builtin (extra int32x8 group before cpol).
__device__ __forceinline__ void tdm_load_tile_16x128_f32(const float* tile_base,
                                                         uint32_t lds_addr) {
  unsigned long long ga = (unsigned long long)(uintptr_t)tile_base;
  v4u g0;
  g0[0] = 1u;                                             // count=1, user desc
  g0[1] = lds_addr;                                       // lds_addr [63:32]
  g0[2] = (uint32_t)ga;                                   // global_addr lo
  g0[3] = (uint32_t)((ga >> 32) & 0x1ffffffull) | (2u << 30); // addr hi | type=2
  v8i g1;
  g1[0] = (int)(2u << 16);                 // data_size=2 (4 bytes), mask=0
  g1[1] = (int)((uint32_t)Tn << 16);       // tensor_dim0 = 128 (lo16)
  g1[2] = (int)((uint32_t)TB << 16);       // dim0 hi=0 | tensor_dim1 = 16 (lo16)
  g1[3] = (int)((uint32_t)Tn << 16);       // dim1 hi=0 | tile_dim0 = 128
  g1[4] = TB;                              // tile_dim1 = 16, tile_dim2 = 0
  g1[5] = Sn * Tn;                         // tensor_dim0_stride = 65536 elems
  g1[6] = 0;                               // stride hi | dim1_stride lo
  g1[7] = 0;
  v4i g2 = {0, 0, 0, 0};
  v4i g3 = {0, 0, 0, 0};
  v8i gx = {0, 0, 0, 0, 0, 0, 0, 0};
  __builtin_amdgcn_tensor_load_to_lds(g0, g1, g2, g3, gx, 0);
}

// ---------------------------------------------------------------------------
// Normalizer: per 16-batch tile, forward recurrence over s with WMMA GEMM.
// ---------------------------------------------------------------------------
__global__ __launch_bounds__(256) void crf_normalizer_kernel(
    const float* __restrict__ emissions,
    const int*   __restrict__ mask,
    const float* __restrict__ start_tr,
    const float* __restrict__ end_tr,
    const float* __restrict__ transitions,
    float* __restrict__ denom)
{
  __shared__ float s_score[TB][Tn + 4];
  __shared__ alignas(16) _Float16 s_p[TB][Tn + 8];   // padded stride: 136 halves
  __shared__ float s_em[2][TB][Tn];                  // TDM double buffer
  __shared__ float s_rowmax[TB];
  __shared__ int   s_maskv[TB];

  const int tid  = threadIdx.x;
  const int wave = tid >> 5;
  const int lane = tid & 31;
  const int lrow = lane & 15;
  const int lhi  = lane >> 4;
  const int b0   = blockIdx.x * TB;

  // Persistent B fragments: expT columns for this wave's N-tile.
  // B 32x16 in this code's contiguous k order (same as A): lane holds fixed N=lrow, K = 16*lhi + h.
  v16h bfrag[4];
#pragma unroll
  for (int c = 0; c < 4; ++c) {
    const int kb = c * 32 + 16 * lhi;
    const int n  = wave * 16 + lrow;
#pragma unroll
    for (int h = 0; h < 16; ++h)
      bfrag[c][h] = (_Float16)__builtin_amdgcn_exp2f(
          transitions[(kb + h) * Tn + n] * LOG2E);
  }

  // score(s=0) = start + emissions[:,0,:]
  for (int idx = tid; idx < TB * Tn; idx += 256) {
    const int r = idx >> 7, j = idx & (Tn - 1);
    s_score[r][j] = start_tr[j] + emissions[((size_t)(b0 + r) * Sn) * Tn + j];
  }

  const uint32_t em_lds_base = lds_off_u32(&s_em[0][0][0]);

  // Prologue: async-load em tile for s=1 into buffer 0.
  if (wave == 0) {
    tdm_load_tile_16x128_f32(emissions + ((size_t)b0 * Sn + 1) * Tn, em_lds_base);
    __builtin_amdgcn_s_wait_tensorcnt(0);
  }
  __syncthreads();

  for (int s = 1; s < Sn; ++s) {
    const int buf = (s - 1) & 1;
    // Prefetch next em tile into the other buffer (wave 0, TDM, async).
    if (wave == 0 && (s + 1) < Sn) {
      tdm_load_tile_16x128_f32(emissions + ((size_t)b0 * Sn + (s + 1)) * Tn,
                               em_lds_base + (uint32_t)(buf ^ 1) * (TB * Tn * 4));
    }

    // Phase 1: per-row max (16 threads per row), and mask load.
    {
      const int r = tid >> 4, g = tid & 15;
      float m = -3.402823e38f;
      for (int j = g; j < Tn; j += 16) m = fmaxf(m, s_score[r][j]);
      m = fmaxf(m, __shfl_xor(m, 8, 32));
      m = fmaxf(m, __shfl_xor(m, 4, 32));
      m = fmaxf(m, __shfl_xor(m, 2, 32));
      m = fmaxf(m, __shfl_xor(m, 1, 32));
      if (g == 0) s_rowmax[r] = m;
      if (tid < TB) s_maskv[tid] = mask[(size_t)(b0 + tid) * Sn + s];
    }
    __syncthreads();

    // Phase 2: p = exp(score - rowmax) as f16  (arg <= 0, raw v_exp_f32 safe).
    {
      const int r = tid >> 4, g = tid & 15;
      const float m = s_rowmax[r];
      for (int j = g; j < Tn; j += 16)
        s_p[r][j] = (_Float16)__builtin_amdgcn_exp2f((s_score[r][j] - m) * LOG2E);
    }
    __syncthreads();

    // Phase 3: Q = P(16x128) x expT(128x128) via 4 chained WMMAs, then update.
    {
      v8f acc = {0.f, 0.f, 0.f, 0.f, 0.f, 0.f, 0.f, 0.f};
      const _Float16* prow = &s_p[lrow][0];
#pragma unroll
      for (int c = 0; c < 4; ++c) {
        // A 16x32 in this code's contiguous k order: lane holds K = 16*lhi + h (matches B)
        const int kc = c * 32 + 16 * lhi;
        h8 lo = *(const h8*)(prow + kc);
        h8 hi = *(const h8*)(prow + kc + 8);
        v16h a;
#pragma unroll
        for (int i = 0; i < 8; ++i) { a[i] = lo[i]; a[i + 8] = hi[i]; }
        acc = __builtin_amdgcn_wmma_f32_16x16x32_f16(
            false, a, false, bfrag[c], (short)0, acc, false, false);
      }
      const int j  = wave * 16 + lrow;   // output column (N)
      const int mb = 8 * lhi;            // C/D layout: VGPR v -> M = mb + v
      const uint32_t em0 =
          em_lds_base + (uint32_t)buf * (TB * Tn * 4) + (uint32_t)((mb * Tn + j) * 4);
      float newsc[8];
#pragma unroll
      for (int v = 0; v < 8; ++v) {
        const int r = mb + v;
        const float em = lds_load_f32_vol(em0 + (uint32_t)v * (Tn * 4));
        // q >= exp(0)*min(expT) ~ 0.9 -> raw v_log_f32 safe.
        const float nv =
            s_rowmax[r] + LN2 * __builtin_amdgcn_logf(acc[v]) + em;
        const float old = s_score[r][j];
        const float mf  = (float)s_maskv[r];
        newsc[v] = old + mf * (nv - old);   // branchless mask select
      }
#pragma unroll
      for (int v = 0; v < 8; ++v) s_score[mb + v][j] = newsc[v];
    }

    // Drain this step's prefetch, then publish new score + next em buffer.
    if (wave == 0 && (s + 1) < Sn) __builtin_amdgcn_s_wait_tensorcnt(0);
    __syncthreads();
  }

  // denom[b] = logsumexp_j(score[b][j] + end_tr[j])
  {
    const int r = tid >> 4, g = tid & 15;
    float m = -3.402823e38f;
    for (int j = g; j < Tn; j += 16) m = fmaxf(m, s_score[r][j] + end_tr[j]);
    m = fmaxf(m, __shfl_xor(m, 8, 32));
    m = fmaxf(m, __shfl_xor(m, 4, 32));
    m = fmaxf(m, __shfl_xor(m, 2, 32));
    m = fmaxf(m, __shfl_xor(m, 1, 32));
    float sum = 0.f;
    for (int j = g; j < Tn; j += 16)
      sum += __builtin_amdgcn_exp2f((s_score[r][j] + end_tr[j] - m) * LOG2E);
    sum += __shfl_xor(sum, 8, 32);
    sum += __shfl_xor(sum, 4, 32);
    sum += __shfl_xor(sum, 2, 32);
    sum += __shfl_xor(sum, 1, 32);
    if (g == 0) denom[b0 + r] = m + LN2 * __builtin_amdgcn_logf(sum);
  }
}

// ---------------------------------------------------------------------------
// Numerator: gold-path score, one wave per batch.
// ---------------------------------------------------------------------------
__global__ __launch_bounds__(256) void crf_score_kernel(
    const float* __restrict__ emissions,
    const int*   __restrict__ mask,
    const int*   __restrict__ tags,
    const float* __restrict__ start_tr,
    const float* __restrict__ end_tr,
    const float* __restrict__ transitions,
    float* __restrict__ numer)
{
  const int wave = threadIdx.x >> 5;
  const int lane = threadIdx.x & 31;
  const int b    = blockIdx.x * 8 + wave;
  float acc = 0.f;
  int msum = 0;
  for (int s = lane; s < Sn; s += 32) {
    const int m  = mask[(size_t)b * Sn + s];
    msum += m;
    const int tg = tags[(size_t)b * Sn + s];
    const float e = emissions[((size_t)b * Sn + s) * Tn + tg];
    float contrib;
    if (s == 0) {
      contrib = start_tr[tg] + e;
    } else {
      const int tp = tags[(size_t)b * Sn + s - 1];
      contrib = (float)m * (transitions[tp * Tn + tg] + e);
    }
    acc += contrib;
  }
  for (int off = 16; off; off >>= 1) {
    acc  += __shfl_xor(acc, off, 32);
    msum += __shfl_xor(msum, off, 32);
  }
  if (lane == 0) {
    const int last = msum - 1;
    numer[b] = acc + end_tr[tags[(size_t)b * Sn + last]];
  }
}

// ---------------------------------------------------------------------------
// Final: out = mean(denom - numer)
// ---------------------------------------------------------------------------
__global__ __launch_bounds__(256) void crf_final_kernel(
    const float* __restrict__ denom, const float* __restrict__ numer,
    float* __restrict__ out)
{
  __shared__ float red[8];
  float v = 0.f;
  for (int i = threadIdx.x; i < Bn; i += 256) v += denom[i] - numer[i];
  for (int off = 16; off; off >>= 1) v += __shfl_xor(v, off, 32);
  const int wave = threadIdx.x >> 5, lane = threadIdx.x & 31;
  if (lane == 0) red[wave] = v;
  __syncthreads();
  if (threadIdx.x == 0) {
    float t = 0.f;
    for (int w = 0; w < 8; ++w) t += red[w];
    out[0] = t / (float)Bn;
  }
}

extern "C" void kernel_launch(void* const* d_in, const int* in_sizes, int n_in,
                              void* d_out, int out_size, void* d_ws, size_t ws_size,
                              hipStream_t stream) {
  (void)in_sizes; (void)n_in; (void)out_size; (void)ws_size;
  const float* emissions = (const float*)d_in[0];
  const int*   mask      = (const int*)d_in[1];
  const int*   tags      = (const int*)d_in[2];
  const float* start_tr  = (const float*)d_in[3];
  const float* end_tr    = (const float*)d_in[4];
  const float* trans     = (const float*)d_in[5];
  float* ws    = (float*)d_ws;
  float* denom = ws;
  float* numer = ws + Bn;

  hipLaunchKernelGGL(crf_normalizer_kernel, dim3(Bn / TB), dim3(256), 0, stream,
                     emissions, mask, start_tr, end_tr, trans, denom);
  hipLaunchKernelGGL(crf_score_kernel, dim3(Bn / 8), dim3(256), 0, stream,
                     emissions, mask, tags, start_tr, end_tr, trans, numer);
  hipLaunchKernelGGL(crf_final_kernel, dim3(1), dim3(256), 0, stream,
                     denom, numer, (float*)d_out);
}
